// GNN_node_57062935495533
// MI455X (gfx1250) — compile-verified, matrix operand order checked by reading, not run on hardware
//
#include <hip/hip_runtime.h>
#include <hip/hip_bf16.h>

typedef __attribute__((ext_vector_type(16))) _Float16 v16h;
typedef __attribute__((ext_vector_type(8)))  float    v8f;

#define NN   50000
#define EE   800000
#define DD   64
#define LL   5
#define BN_EPS 1e-5f

// ---------------------------------------------------------------- utilities
__global__ void zero_kernel(float* p, int n) {
    int i = blockIdx.x * blockDim.x + threadIdx.x;
    if (i < n) p[i] = 0.0f;
}

__global__ void init_h_kernel(const int* __restrict__ x,
                              const float* __restrict__ emb,
                              float* __restrict__ h) {
    int i = blockIdx.x * blockDim.x + threadIdx.x;
    if (i < NN * DD) {
        int node = i >> 6;
        int d    = i & 63;
        h[i] = emb[x[node] * DD + d];
    }
}

// ------------------------------------------------- edge encode + gather + scatter
// one wave32 per edge; lane t owns dims 2t, 2t+1
__global__ __launch_bounds__(256)
void edge_kernel(const int* __restrict__ ei, const float* __restrict__ eattr,
                 const float* __restrict__ We, const float* __restrict__ be,
                 int layer, const float* __restrict__ h, float* __restrict__ agg) {
    __shared__ float sWe[7 * DD];
    __shared__ float sbe[DD];
    for (int i = threadIdx.x; i < 7 * DD; i += blockDim.x) sWe[i] = We[layer * 7 * DD + i];
    for (int i = threadIdx.x; i < DD;     i += blockDim.x) sbe[i] = be[layer * DD + i];
    __syncthreads();

    int lane   = threadIdx.x & 31;
    int wave   = blockIdx.x * (blockDim.x >> 5) + (threadIdx.x >> 5);
    int nwaves = gridDim.x * (blockDim.x >> 5);
    int d0 = lane * 2;

    for (int e = wave; e < EE; e += nwaves) {
        // prefetch next iteration's edge attributes (global_prefetch_b8)
        if (e + nwaves < EE)
            __builtin_prefetch(&eattr[(size_t)(e + nwaves) * 7], 0, 1);

        float av = (lane < 7) ? eattr[e * 7 + lane] : 0.0f;
        int src = ei[e];
        int dst = ei[EE + e];
        float m0 = sbe[d0];
        float m1 = sbe[d0 + 1];
#pragma unroll
        for (int j = 0; j < 7; ++j) {
            float aj = __shfl(av, j, 32);
            m0 = fmaf(aj, sWe[j * DD + d0],     m0);
            m1 = fmaf(aj, sWe[j * DD + d0 + 1], m1);
        }
        const float2 hv = *reinterpret_cast<const float2*>(&h[src * DD + d0]);
        m0 = fmaxf(m0 + hv.x, 0.0f);
        m1 = fmaxf(m1 + hv.y, 0.0f);
        atomicAdd(&agg[dst * DD + d0],     m0);
        atomicAdd(&agg[dst * DD + d0 + 1], m1);
    }
}

// ------------------------------------------------- GEMM1: y1 = ((1+eps)h + agg) @ W1 + b1
// wave-per-16-row tile, 8 N-tiles (128 cols), K = 64 -> 2 chunks of 32, f16 WMMA f32-acc.
// W1 is staged into LDS PRE-SWIZZLED into per-lane WMMA B-fragment order so each
// lane's 16 halves are one contiguous 32B run -> ds_load_b128 pairs.
__global__ __launch_bounds__(256)
void gemm1_kernel(const float* __restrict__ h, const float* __restrict__ agg,
                  const float* __restrict__ W1, const float* __restrict__ b1,
                  const float* __restrict__ epsv, int layer,
                  float* __restrict__ y1, float* __restrict__ colsum,
                  float* __restrict__ colsq) {
    __shared__ __align__(16) _Float16 sW[DD * 128];   // 16 KB, swizzled
    __shared__ float sb[128];
    // swizzled stage: p = ((kc*8 + nt)*32 + lane)*16 + i
    for (int p = threadIdx.x; p < DD * 128; p += blockDim.x) {
        int i    = p & 15;
        int ln   = (p >> 4) & 31;
        int nt   = (p >> 9) & 7;
        int kc   = p >> 12;
        int khp  = ln >> 4;
        int col  = nt * 16 + (ln & 15);
        int kk   = kc * 32 + ((i < 8) ? (khp * 8 + i) : (16 + khp * 8 + (i - 8)));
        sW[p] = (_Float16)W1[layer * DD * 128 + kk * 128 + col];
    }
    for (int i = threadIdx.x; i < 128; i += blockDim.x)
        sb[i] = b1[layer * 128 + i];
    __syncthreads();

    float epsl = 1.0f + epsv[layer];
    int lane = threadIdx.x & 31;
    int wave = blockIdx.x * (blockDim.x >> 5) + (threadIdx.x >> 5);
    int nw   = gridDim.x * (blockDim.x >> 5);
    int r  = lane & 15;
    int kh = lane >> 4;
    const int nTiles = NN / 16;   // 3125

    for (int tile = wave; tile < nTiles; tile += nw) {
        v8f c[8];
#pragma unroll
        for (int nt = 0; nt < 8; ++nt) c[nt] = (v8f){0,0,0,0,0,0,0,0};

        int row = tile * 16 + r;
#pragma unroll
        for (int kc = 0; kc < 2; ++kc) {
            int k0 = kc * 32;
            v16h a;
#pragma unroll
            for (int i = 0; i < 8; ++i) {
                int ka = k0 + kh * 8 + i;
                int kb = k0 + 16 + kh * 8 + i;
                a[i]     = (_Float16)fmaf(epsl, h[row * DD + ka], agg[row * DD + ka]);
                a[i + 8] = (_Float16)fmaf(epsl, h[row * DD + kb], agg[row * DD + kb]);
            }
#pragma unroll
            for (int nt = 0; nt < 8; ++nt) {
                v16h bm = *reinterpret_cast<const v16h*>(
                              &sW[((kc * 8 + nt) * 32 + lane) * 16]);
                c[nt] = __builtin_amdgcn_wmma_f32_16x16x32_f16(
                            false, a, false, bm, (short)0, c[nt], false, false);
            }
        }
        // epilogue: +bias, store, per-column batch stats
        int rowBase = tile * 16 + kh * 8;
#pragma unroll
        for (int nt = 0; nt < 8; ++nt) {
            int col = nt * 16 + r;
            float s = 0.0f, sq = 0.0f;
#pragma unroll
            for (int v = 0; v < 8; ++v) {
                float val = c[nt][v] + sb[col];
                y1[(rowBase + v) * 128 + col] = val;
                s  += val;
                sq += val * val;
            }
            s  += __shfl_xor(s, 16, 32);
            sq += __shfl_xor(sq, 16, 32);
            if (kh == 0) {
                atomicAdd(&colsum[col], s);
                atomicAdd(&colsq[col], sq);
            }
        }
    }
}

// ------------------------------------------------- GEMM2: y2 = relu(BN(y1)) @ W2 + b2
__global__ __launch_bounds__(256)
void gemm2_kernel(const float* __restrict__ y1,
                  const float* __restrict__ alpha1, const float* __restrict__ beta1,
                  const float* __restrict__ W2, const float* __restrict__ b2,
                  int layer, float* __restrict__ y2,
                  float* __restrict__ colsum, float* __restrict__ colsq) {
    __shared__ __align__(16) _Float16 sW[128 * DD];   // 16 KB, swizzled
    __shared__ float sA[128];
    __shared__ float sB[128];
    __shared__ float sb[DD];
    // swizzled stage: p = ((kc*4 + nt)*32 + lane)*16 + i, kc in 0..3, nt in 0..3
    for (int p = threadIdx.x; p < 128 * DD; p += blockDim.x) {
        int i    = p & 15;
        int ln   = (p >> 4) & 31;
        int nt   = (p >> 9) & 3;
        int kc   = (p >> 11) & 3;
        int khp  = ln >> 4;
        int col  = nt * 16 + (ln & 15);
        int kk   = kc * 32 + ((i < 8) ? (khp * 8 + i) : (16 + khp * 8 + (i - 8)));
        sW[p] = (_Float16)W2[layer * 128 * DD + kk * DD + col];
    }
    for (int i = threadIdx.x; i < 128; i += blockDim.x) { sA[i] = alpha1[i]; sB[i] = beta1[i]; }
    for (int i = threadIdx.x; i < DD;  i += blockDim.x) sb[i] = b2[layer * DD + i];
    __syncthreads();

    int lane = threadIdx.x & 31;
    int wave = blockIdx.x * (blockDim.x >> 5) + (threadIdx.x >> 5);
    int nw   = gridDim.x * (blockDim.x >> 5);
    int r  = lane & 15;
    int kh = lane >> 4;
    const int nTiles = NN / 16;

    for (int tile = wave; tile < nTiles; tile += nw) {
        v8f c[4];
#pragma unroll
        for (int nt = 0; nt < 4; ++nt) c[nt] = (v8f){0,0,0,0,0,0,0,0};

        int row = tile * 16 + r;
#pragma unroll
        for (int kc = 0; kc < 4; ++kc) {
            int k0 = kc * 32;
            v16h a;
#pragma unroll
            for (int i = 0; i < 8; ++i) {
                int ka = k0 + kh * 8 + i;
                int kb = k0 + 16 + kh * 8 + i;
                a[i]     = (_Float16)fmaxf(fmaf(y1[row * 128 + ka], sA[ka], sB[ka]), 0.0f);
                a[i + 8] = (_Float16)fmaxf(fmaf(y1[row * 128 + kb], sA[kb], sB[kb]), 0.0f);
            }
#pragma unroll
            for (int nt = 0; nt < 4; ++nt) {
                v16h bm = *reinterpret_cast<const v16h*>(
                              &sW[((kc * 4 + nt) * 32 + lane) * 16]);
                c[nt] = __builtin_amdgcn_wmma_f32_16x16x32_f16(
                            false, a, false, bm, (short)0, c[nt], false, false);
            }
        }
        int rowBase = tile * 16 + kh * 8;
#pragma unroll
        for (int nt = 0; nt < 4; ++nt) {
            int col = nt * 16 + r;
            float s = 0.0f, sq = 0.0f;
#pragma unroll
            for (int v = 0; v < 8; ++v) {
                float val = c[nt][v] + sb[col];
                y2[(rowBase + v) * DD + col] = val;
                s  += val;
                sq += val * val;
            }
            s  += __shfl_xor(s, 16, 32);
            sq += __shfl_xor(sq, 16, 32);
            if (kh == 0) {
                atomicAdd(&colsum[col], s);
                atomicAdd(&colsq[col], sq);
            }
        }
    }
}

// ------------------------------------------------- BN finalize: alpha/beta from stats
__global__ void bnfin_kernel(const float* __restrict__ cs, const float* __restrict__ cq,
                             const float* __restrict__ g, const float* __restrict__ bb,
                             int layer, int C, float* __restrict__ alpha,
                             float* __restrict__ beta) {
    int n = threadIdx.x;
    if (n < C) {
        float m   = cs[n] * (1.0f / NN);
        float var = cq[n] * (1.0f / NN) - m * m;
        float inv = rsqrtf(var + BN_EPS);
        float a   = g[layer * C + n] * inv;
        alpha[n] = a;
        beta[n]  = bb[layer * C + n] - m * a;
    }
}

// ------------------------------------------------- apply outer BN (+optional ReLU)
__global__ void bnapply_kernel(const float* __restrict__ y2,
                               const float* __restrict__ alpha, const float* __restrict__ beta,
                               int doRelu, float* __restrict__ out) {
    int i = blockIdx.x * blockDim.x + threadIdx.x;
    if (i < NN * DD) {
        int col = i & 63;
        float v = fmaf(y2[i], alpha[col], beta[col]);
        if (doRelu) v = fmaxf(v, 0.0f);
        out[i] = v;
    }
}

// ---------------------------------------------------------------- host launcher
extern "C" void kernel_launch(void* const* d_in, const int* in_sizes, int n_in,
                              void* d_out, int out_size, void* d_ws, size_t ws_size,
                              hipStream_t stream) {
    const int*   x     = (const int*)  d_in[0];
    const int*   ei    = (const int*)  d_in[1];
    const float* eattr = (const float*)d_in[2];
    const float* emb   = (const float*)d_in[3];
    const float* We    = (const float*)d_in[4];
    const float* be    = (const float*)d_in[5];
    const float* epsv  = (const float*)d_in[6];
    const float* W1    = (const float*)d_in[7];
    const float* b1    = (const float*)d_in[8];
    const float* g1    = (const float*)d_in[9];
    const float* bb1   = (const float*)d_in[10];
    const float* W2    = (const float*)d_in[11];
    const float* b2    = (const float*)d_in[12];
    const float* g2    = (const float*)d_in[13];
    const float* bb2   = (const float*)d_in[14];

    float* ws   = (float*)d_ws;
    float* h    = ws;                       // N*64
    float* agg  = h   + (size_t)NN * DD;    // N*64  (reused as y2)
    float* y1   = agg + (size_t)NN * DD;    // N*128
    float* st   = y1  + (size_t)NN * 128;   // stats + affine params
    float* cs1 = st;        float* cq1 = st + 128;
    float* cs2 = st + 256;  float* cq2 = st + 320;   // 384 floats to zero
    float* al1 = st + 384;  float* be1 = st + 512;
    float* al2 = st + 640;  float* be2 = st + 704;
    float* y2  = agg;                        // alias: agg dead after gemm1

    const int elemBlocks = (NN * DD + 255) / 256;

    init_h_kernel<<<elemBlocks, 256, 0, stream>>>(x, emb, h);

    for (int l = 0; l < LL; ++l) {
        zero_kernel<<<elemBlocks, 256, 0, stream>>>(agg, NN * DD);
        zero_kernel<<<2, 256, 0, stream>>>(st, 384);

        edge_kernel<<<1024, 256, 0, stream>>>(ei, eattr, We, be, l, h, agg);

        gemm1_kernel<<<391, 256, 0, stream>>>(h, agg, W1, b1, epsv, l, y1, cs1, cq1);
        bnfin_kernel<<<1, 128, 0, stream>>>(cs1, cq1, g1, bb1, l, 128, al1, be1);

        gemm2_kernel<<<391, 256, 0, stream>>>(y1, al1, be1, W2, b2, l, y2, cs2, cq2);
        bnfin_kernel<<<1, 128, 0, stream>>>(cs2, cq2, g2, bb2, l, DD, al2, be2);

        float* dst = (l == LL - 1) ? (float*)d_out : h;
        bnapply_kernel<<<elemBlocks, 256, 0, stream>>>(y2, al2, be2, (l < LL - 1) ? 1 : 0, dst);
    }
}
